// CondAttLSTM_5463198400974
// MI455X (gfx1250) — matrix operand order repaired and checked by reading, not compiled, not run on hardware
//
#include <hip/hip_runtime.h>
#include <hip/hip_bf16.h>

typedef __bf16 bf16_t;
typedef __attribute__((ext_vector_type(16))) __bf16 v16bf;
typedef __attribute__((ext_vector_type(8)))  __bf16 v8bf;
typedef __attribute__((ext_vector_type(8)))  float  v8f;

// ---------------- problem constants ----------------
constexpr int Bc = 16, Tc = 128, Sc = 128, Dc = 256, Hc = 256, Cc = 256, Ac = 256;
constexpr int COMBc = Hc + Cc + Hc + Hc;   // 1024
constexpr float NEGV = -1e9f;

// ---------------- workspace layout (f32 region then bf16 region) ----------------
constexpr size_t N_BTH   = (size_t)Bc * Tc * Hc;          // 524288
constexpr size_t OFF_XI  = 0;
constexpr size_t OFF_XF  = OFF_XI  + N_BTH;
constexpr size_t OFF_XO  = OFF_XF  + N_BTH;
constexpr size_t OFF_XC  = OFF_XO  + N_BTH;
constexpr size_t OFF_CTXT= OFF_XC  + N_BTH;               // ctx_trans [B,S,A]
constexpr size_t OFF_HIST= OFF_CTXT+ N_BTH;               // hist      [B,T,H]
constexpr size_t OFF_HHB = OFF_HIST+ N_BTH;               // HH cache  [B,T,A]
constexpr size_t OFF_HST = OFF_HHB + N_BTH;               // h state   [B,H]
constexpr size_t OFF_CST = OFF_HST + (size_t)Bc * Hc;     // c state   [B,H]
constexpr size_t F32_TOTAL = OFF_CST + (size_t)Bc * Hc;

constexpr size_t BF_XBF   = 0;                               // X bf16 [B,T,D]
constexpr size_t BF_CTX   = BF_XBF + (size_t)Bc * Tc * Dc;   // context bf16 [B,S,C]
constexpr size_t BF_WIX   = BF_CTX + (size_t)Bc * Sc * Cc;
constexpr size_t BF_WFX   = BF_WIX   + (size_t)Hc * Dc;
constexpr size_t BF_WOX   = BF_WFX   + (size_t)Hc * Dc;
constexpr size_t BF_WCX   = BF_WOX   + (size_t)Hc * Dc;
constexpr size_t BF_WACTX = BF_WCX   + (size_t)Hc * Dc;
constexpr size_t BF_WATH  = BF_WACTX + (size_t)Ac * Cc;
constexpr size_t BF_WHATH = BF_WATH  + (size_t)Ac * Hc;
constexpr size_t BF_WHHIST= BF_WHATH + (size_t)Ac * Hc;
constexpr size_t BF_WI    = BF_WHHIST+ (size_t)Ac * Hc;
constexpr size_t BF_WF    = BF_WI    + (size_t)Hc * COMBc;
constexpr size_t BF_WC    = BF_WF    + (size_t)Hc * COMBc;
constexpr size_t BF_WO    = BF_WC    + (size_t)Hc * COMBc;

// ---------------- WMMA fragment helpers (16x16x32 bf16, wave32) ----------------
// A fragment: 16x32 (MxK). lanes 0-15 / 16-31 hold the same M rows, different K
// halves: elems 0..7 -> K = hf*8 + e ; elems 8..15 -> K = 16 + hf*8 + (e-8).
__device__ inline v16bf load_a_frag(const bf16_t* p, int ld, int m0, int k0, int lane) {
  int hf = lane >> 4, m = lane & 15;
  const bf16_t* row = p + (size_t)(m0 + m) * ld + k0 + hf * 8;
  v8bf lo = *(const v8bf*)(row);
  v8bf hi = *(const v8bf*)(row + 16);
  return __builtin_shufflevector(lo, hi, 0,1,2,3,4,5,6,7,8,9,10,11,12,13,14,15);
}

// B fragment: 32x16 (KxN); since every GEMM here is Y = X @ W.T, B[k,n] = W[n,k],
// i.e. B column n is W row n; same K order as A: elems 0..7 -> K = hf*8 + e ; elems 8..15 -> K = 16 + hf*8 + (e-8).
__device__ inline v16bf load_b_frag(const bf16_t* p, int ld, int n0, int k0, int lane) {
  int hf = lane >> 4, n = lane & 15;
  const bf16_t* col = p + (size_t)(n0 + n) * ld + k0 + hf * 8;
  v8bf lo = *(const v8bf*)(col);
  v8bf hi = *(const v8bf*)(col + 16);
  return __builtin_shufflevector(lo, hi, 0,1,2,3,4,5,6,7,8,9,10,11,12,13,14,15);
}

__device__ inline v8f wmma_bf16(v16bf a, v16bf b, v8f c) {
  return __builtin_amdgcn_wmma_f32_16x16x32_bf16(false, a, false, b, (short)0, c, false, false);
}

__device__ inline float sigm(float x) { return 1.0f / (1.0f + __expf(-x)); }

// Wave32 softmax over a 128-wide LDS row (4 values per lane), in place.
__device__ inline void wave_softmax_row128(float* row, int lane) {
  float v[4]; float mx = -3.0e38f;
  #pragma unroll
  for (int i = 0; i < 4; ++i) { v[i] = row[lane + 32 * i]; mx = fmaxf(mx, v[i]); }
  #pragma unroll
  for (int m = 16; m >= 1; m >>= 1) mx = fmaxf(mx, __shfl_xor(mx, m, 32));
  float sum = 0.0f;
  #pragma unroll
  for (int i = 0; i < 4; ++i) { v[i] = __expf(v[i] - mx); sum += v[i]; }
  #pragma unroll
  for (int m = 16; m >= 1; m >>= 1) sum += __shfl_xor(sum, m, 32);
  float inv = 1.0f / sum;
  #pragma unroll
  for (int i = 0; i < 4; ++i) row[lane + 32 * i] = v[i] * inv;
}

// ---------------- precompute kernels ----------------
__global__ void k_f2bf(const float* __restrict__ src, bf16_t* __restrict__ dst, int n) {
  int i = blockIdx.x * blockDim.x + threadIdx.x;
  int stride = gridDim.x * blockDim.x;
  for (; i < n; i += stride) dst[i] = (bf16_t)src[i];
}

// out[M,N] = A[M,K](bf16, row-major) @ W[N,K]^T (bf16) + bias[N]; one wave per 16x16 tile.
__global__ __launch_bounds__(256)
void k_gemm_bias(const bf16_t* __restrict__ Am, const bf16_t* __restrict__ Wm,
                 const float* __restrict__ bias, float* __restrict__ out,
                 int M, int N, int K) {
  int lane = threadIdx.x & 31;
  int wid  = blockIdx.x * (blockDim.x >> 5) + (threadIdx.x >> 5);
  int ntiles = N >> 4;
  if (wid >= (M >> 4) * ntiles) return;
  int m0 = (wid / ntiles) << 4, n0 = (wid % ntiles) << 4;
  v8f acc = {};
  for (int k0 = 0; k0 < K; k0 += 32) {
    __builtin_prefetch(Wm + (size_t)(n0 + (lane & 15)) * K + k0 + 64, 0, 1);
    v16bf a = load_a_frag(Am, K, m0, k0, lane);
    v16bf b = load_b_frag(Wm, K, n0, k0, lane);
    acc = wmma_bf16(a, b, acc);
  }
  int hf = lane >> 4, l16 = lane & 15;
  float bv = bias ? bias[n0 + l16] : 0.0f;
  #pragma unroll
  for (int r = 0; r < 8; ++r)
    out[(size_t)(m0 + r + 8 * hf) * N + n0 + l16] = acc[r] + bv;
}

// ---------------- fused per-timestep kernel (1 workgroup, 16 waves) ----------------
__global__ __launch_bounds__(512)
void k_step(int t,
            const float* __restrict__ context,   // [B,S,C] f32
            const int*   __restrict__ parent_t,  // [T]
            const float* __restrict__ W_att, const float* __restrict__ b_att,
            const float* __restrict__ b_hhist,
            const float* __restrict__ W_hatt, const float* __restrict__ b_hatt,
            float* __restrict__ wsf, const bf16_t* __restrict__ wsb,
            float* __restrict__ out_h, float* __restrict__ out_ctx) {
  const float* Xi   = wsf + OFF_XI;
  const float* Xf   = wsf + OFF_XF;
  const float* Xo   = wsf + OFF_XO;
  const float* Xc   = wsf + OFF_XC;
  const float* ctxt = wsf + OFF_CTXT;
  float* hist = wsf + OFF_HIST;
  float* HHc  = wsf + OFF_HHB;
  float* hst  = wsf + OFF_HST;
  float* cst  = wsf + OFF_CST;
  const bf16_t* Wath_b   = wsb + BF_WATH;
  const bf16_t* Whath_b  = wsb + BF_WHATH;
  const bf16_t* Whhist_b = wsb + BF_WHHIST;
  const bf16_t* Wi_b = wsb + BF_WI;
  const bf16_t* Wf_b = wsb + BF_WF;
  const bf16_t* Wc_b = wsb + BF_WC;
  const bf16_t* Wo_b = wsb + BF_WO;

  __shared__ __align__(16) bf16_t sh_comb[16][1032];  // h_comb bf16 (padded rows)
  __shared__ __align__(16) float  sh_tmp[16][260];    // aW / hW ; reused as bf16 h_new
  __shared__ __align__(16) float  sh_sc[16][132];     // scores / softmax weights
  bf16_t* sh_hnew = (bf16_t*)&sh_tmp[0][0];           // [16][264] bf16 overlay (phase P6+)

  int tid = threadIdx.x, lane = tid & 31, wave = tid >> 5;
  int hf = lane >> 4, l16 = lane & 15;

  // ---- P0: h -> bf16 seg0 ; parent hidden -> seg2 ; zero h_ctx at t==0 ----
  int pt = parent_t[t];
  for (int idx = tid; idx < Bc * Hc; idx += 512) {
    int b = idx >> 8, j = idx & 255;
    sh_comb[b][j] = (bf16_t)hst[idx];
    float pv = (t > 0) ? hist[((size_t)b * Tc + pt) * Hc + j] : 0.0f;
    sh_comb[b][512 + j] = (bf16_t)pv;
    if (t == 0) sh_comb[b][768 + j] = (bf16_t)0.0f;
  }
  __syncthreads();

  // ---- P1: aW = h @ W_ath^T  (wave w -> N-tile w) ----
  {
    int n0 = wave << 4;
    v8f acc = {};
    for (int k0 = 0; k0 < Hc; k0 += 32) {
      v16bf a = load_a_frag(&sh_comb[0][0], 1032, 0, k0, lane);
      v16bf b = load_b_frag(Wath_b, Hc, n0, k0, lane);
      acc = wmma_bf16(a, b, acc);
    }
    #pragma unroll
    for (int r = 0; r < 8; ++r) sh_tmp[r + 8 * hf][n0 + l16] = acc[r];
  }
  __syncthreads();

  // ---- P2: context attention ----
  for (int idx = tid; idx < Bc * Sc; idx += 512) {
    int b = idx >> 7, s = idx & 127;
    const float* ct = ctxt + ((size_t)b * Sc + s) * Ac;
    float acc = 0.0f;
    for (int a = 0; a < Ac; ++a) acc += tanhf(ct[a] + sh_tmp[b][a]) * W_att[a];
    sh_sc[b][s] = acc + b_att[0];
  }
  __syncthreads();
  wave_softmax_row128(sh_sc[wave], lane);   // 16 waves == 16 batch rows
  __syncthreads();
  for (int idx = tid; idx < Bc * Cc; idx += 512) {
    int b = idx >> 8, cd = idx & 255;
    float acc = 0.0f;
    for (int s = 0; s < Sc; ++s) acc += sh_sc[b][s] * context[((size_t)b * Sc + s) * Cc + cd];
    out_ctx[((size_t)t * Bc + b) * Cc + cd] = acc;
    sh_comb[b][256 + cd] = (bf16_t)acc;
  }
  __syncthreads();

  if (t > 0) {
    // ---- P3: hW = h @ W_hath^T ----
    {
      int n0 = wave << 4;
      v8f acc = {};
      for (int k0 = 0; k0 < Hc; k0 += 32) {
        v16bf a = load_a_frag(&sh_comb[0][0], 1032, 0, k0, lane);
        v16bf b = load_b_frag(Whath_b, Hc, n0, k0, lane);
        acc = wmma_bf16(a, b, acc);
      }
      #pragma unroll
      for (int r = 0; r < 8; ++r) sh_tmp[r + 8 * hf][n0 + l16] = acc[r];
    }
    __syncthreads();
    // ---- P4: history attention using incrementally-maintained HH cache ----
    for (int idx = tid; idx < Bc * Tc; idx += 512) {
      int b = idx >> 7, j = idx & 127;
      float acc = NEGV;
      if (j < t) {
        const float* hh = HHc + ((size_t)b * Tc + j) * Ac;
        acc = b_hatt[0];
        for (int a = 0; a < Ac; ++a) acc += tanhf(hh[a] + b_hhist[a] + sh_tmp[b][a]) * W_hatt[a];
      }
      sh_sc[b][j] = acc;
    }
    __syncthreads();
    wave_softmax_row128(sh_sc[wave], lane);
    __syncthreads();
    for (int idx = tid; idx < Bc * Hc; idx += 512) {
      int b = idx >> 8, hd = idx & 255;
      float acc = 0.0f;
      for (int j = 0; j < t; ++j) acc += sh_sc[b][j] * hist[((size_t)b * Tc + j) * Hc + hd];
      sh_comb[b][768 + hd] = (bf16_t)acc;
    }
    __syncthreads();
  }

  // ---- P6: four gate GEMMs, A-fragment reused across i/f/c/o, + LSTM cell ----
  {
    int n0 = wave << 4;
    v8f ai = {}, af = {}, ag = {}, ao = {};
    for (int k0 = 0; k0 < COMBc; k0 += 32) {
      v16bf a  = load_a_frag(&sh_comb[0][0], 1032, 0, k0, lane);
      v16bf bi = load_b_frag(Wi_b, COMBc, n0, k0, lane); ai = wmma_bf16(a, bi, ai);
      v16bf bf = load_b_frag(Wf_b, COMBc, n0, k0, lane); af = wmma_bf16(a, bf, af);
      v16bf bc = load_b_frag(Wc_b, COMBc, n0, k0, lane); ag = wmma_bf16(a, bc, ag);
      v16bf bo = load_b_frag(Wo_b, COMBc, n0, k0, lane); ao = wmma_bf16(a, bo, ao);
    }
    #pragma unroll
    for (int r = 0; r < 8; ++r) {
      int b = r + 8 * hf, hd = n0 + l16;
      size_t gidx = ((size_t)b * Tc + t) * Hc + hd;
      size_t sidx = (size_t)b * Hc + hd;
      float iv = sigm(Xi[gidx] + ai[r]);
      float fv = sigm(Xf[gidx] + af[r]);
      float gv = tanhf(Xc[gidx] + ag[r]);
      float ov = sigm(Xo[gidx] + ao[r]);
      float cn = fv * cst[sidx] + iv * gv;
      cst[sidx] = cn;
      float hn = ov * tanhf(cn);
      hist[gidx] = hn;
      hst[sidx]  = hn;
      out_h[gidx] = hn;
      sh_hnew[b * 264 + hd] = (bf16_t)hn;
    }
  }
  __syncthreads();

  // ---- P7: HH[:,t,:] = h_new @ W_hhist^T  (incremental history-projection) ----
  {
    int n0 = wave << 4;
    v8f acc = {};
    for (int k0 = 0; k0 < Hc; k0 += 32) {
      v16bf a = load_a_frag(sh_hnew, 264, 0, k0, lane);
      v16bf b = load_b_frag(Whhist_b, Hc, n0, k0, lane);
      acc = wmma_bf16(a, b, acc);
    }
    #pragma unroll
    for (int r = 0; r < 8; ++r)
      HHc[((size_t)(r + 8 * hf) * Tc + t) * Ac + n0 + l16] = acc[r];
  }
}

// ---------------- host launch ----------------
extern "C" void kernel_launch(void* const* d_in, const int* in_sizes, int n_in,
                              void* d_out, int out_size, void* d_ws, size_t ws_size,
                              hipStream_t stream) {
  (void)in_sizes; (void)n_in; (void)out_size; (void)ws_size;
  const float* X       = (const float*)d_in[0];
  const float* context = (const float*)d_in[1];
  const float* h0      = (const float*)d_in[2];
  const float* c0      = (const float*)d_in[3];
  const int*   parentT = (const int*)d_in[4];
  const float* W_ix = (const float*)d_in[5];  const float* b_ix = (const float*)d_in[6];
  const float* W_fx = (const float*)d_in[7];  const float* b_fx = (const float*)d_in[8];
  const float* W_ox = (const float*)d_in[9];  const float* b_ox = (const float*)d_in[10];
  const float* W_cx = (const float*)d_in[11]; const float* b_cx = (const float*)d_in[12];
  const float* W_i  = (const float*)d_in[13];
  const float* W_f  = (const float*)d_in[14];
  const float* W_c  = (const float*)d_in[15];
  const float* W_o  = (const float*)d_in[16];
  const float* W_actx = (const float*)d_in[17]; const float* b_actx = (const float*)d_in[18];
  const float* W_ath  = (const float*)d_in[19];
  const float* W_att  = (const float*)d_in[20]; const float* b_att  = (const float*)d_in[21];
  const float* W_hhist= (const float*)d_in[22]; const float* b_hhist= (const float*)d_in[23];
  const float* W_hath = (const float*)d_in[24];
  const float* W_hatt = (const float*)d_in[25]; const float* b_hatt = (const float*)d_in[26];

  float*  wsf = (float*)d_ws;
  bf16_t* wsb = (bf16_t*)((char*)d_ws + F32_TOTAL * sizeof(float));

  // LSTM carried state
  hipMemcpyAsync(wsf + OFF_HST, h0, (size_t)Bc * Hc * sizeof(float), hipMemcpyDeviceToDevice, stream);
  hipMemcpyAsync(wsf + OFF_CST, c0, (size_t)Bc * Hc * sizeof(float), hipMemcpyDeviceToDevice, stream);

  auto cvt = [&](const float* s, size_t off, int n) {
    int nb = (n + 1023) >> 10; if (nb < 1) nb = 1;
    k_f2bf<<<nb, 256, 0, stream>>>(s, wsb + off, n);
  };
  cvt(X,       BF_XBF,   Bc * Tc * Dc);
  cvt(context, BF_CTX,   Bc * Sc * Cc);
  cvt(W_ix,    BF_WIX,   Hc * Dc);
  cvt(W_fx,    BF_WFX,   Hc * Dc);
  cvt(W_ox,    BF_WOX,   Hc * Dc);
  cvt(W_cx,    BF_WCX,   Hc * Dc);
  cvt(W_actx,  BF_WACTX, Ac * Cc);
  cvt(W_ath,   BF_WATH,  Ac * Hc);
  cvt(W_hath,  BF_WHATH, Ac * Hc);
  cvt(W_hhist, BF_WHHIST,Ac * Hc);
  cvt(W_i,     BF_WI,    Hc * COMBc);
  cvt(W_f,     BF_WF,    Hc * COMBc);
  cvt(W_c,     BF_WC,    Hc * COMBc);
  cvt(W_o,     BF_WO,    Hc * COMBc);

  // Precomputed projections: Xi/Xf/Xo/Xc [B*T,H], ctx_trans [B*S,A]
  k_gemm_bias<<<256, 256, 0, stream>>>(wsb + BF_XBF, wsb + BF_WIX, b_ix, wsf + OFF_XI,  Bc*Tc, Hc, Dc);
  k_gemm_bias<<<256, 256, 0, stream>>>(wsb + BF_XBF, wsb + BF_WFX, b_fx, wsf + OFF_XF,  Bc*Tc, Hc, Dc);
  k_gemm_bias<<<256, 256, 0, stream>>>(wsb + BF_XBF, wsb + BF_WOX, b_ox, wsf + OFF_XO,  Bc*Tc, Hc, Dc);
  k_gemm_bias<<<256, 256, 0, stream>>>(wsb + BF_XBF, wsb + BF_WCX, b_cx, wsf + OFF_XC,  Bc*Tc, Hc, Dc);
  k_gemm_bias<<<256, 256, 0, stream>>>(wsb + BF_CTX, wsb + BF_WACTX, b_actx, wsf + OFF_CTXT, Bc*Sc, Ac, Cc);

  float* out_h   = (float*)d_out;
  float* out_ctx = out_h + (size_t)Bc * Tc * Hc;
  for (int t = 0; t < Tc; ++t)
    k_step<<<1, 512, 0, stream>>>(t, context, parentT, W_att, b_att, b_hhist,
                                  W_hatt, b_hatt, wsf, wsb, out_h, out_ctx);
}
